// PNAPCSAFT_10479720202990
// MI455X (gfx1250) — matrix operand order repaired and checked by reading, not compiled, not run on hardware
//
#include <hip/hip_runtime.h>
#include <hip/hip_bf16.h>
#include <math.h>

#define Hh   128
#define NL   2
#define NN   40000
#define NE   640000
#define NET  (NE + NN)   /* 680000, divisible by 64; NE divisible by 64 too */
#define NG   1000
#define NP   3

typedef __attribute__((ext_vector_type(16))) __bf16         v16bf;
typedef __attribute__((ext_vector_type(8)))  float          v8f;
typedef __attribute__((ext_vector_type(8)))  unsigned short us8v;
typedef __attribute__((ext_vector_type(16))) unsigned short us16v;
typedef __attribute__((ext_vector_type(4)))  int            v4i;

// ---- CDNA5 async global->LDS path (guarded; falls back to plain copies) ----
#if defined(__has_builtin)
#  if __has_builtin(__builtin_amdgcn_global_load_async_to_lds_b128)
#    define HAVE_ASYNC_LDS 1
#  endif
#endif
#ifndef HAVE_ASYNC_LDS
#  define HAVE_ASYNC_LDS 0
#endif

#if HAVE_ASYNC_LDS
typedef __attribute__((address_space(1))) v4i gv4i;   // global v4i32
typedef __attribute__((address_space(3))) v4i lv4i;   // LDS v4i32
#  if __has_builtin(__builtin_amdgcn_s_wait_asynccnt)
#    define WAIT_ASYNC(n) __builtin_amdgcn_s_wait_asynccnt(n)
#  else
#    define WAIT_ASYNC(n) asm volatile("s_wait_asynccnt %0" ::"i"(n) : "memory")
#  endif
#else
#  define WAIT_ASYNC(n) ((void)0)
#endif

__device__ __forceinline__ unsigned short f2bf(float f) {
  unsigned int u = __builtin_bit_cast(unsigned int, f);
  unsigned int r = (u + 0x7FFFu + ((u >> 16) & 1u)) >> 16;  // RNE
  return (unsigned short)r;
}

__device__ __forceinline__ v16bf mk16(us8v lo, us8v hi) {
  us16v t;
#pragma unroll
  for (int i = 0; i < 8; i++) { t[i] = lo[i]; t[i + 8] = hi[i]; }
  return __builtin_bit_cast(v16bf, t);
}

__device__ __forceinline__ v8f wmma_bf16(v16bf a, v16bf b, v8f c) {
  // (neg_a, A, neg_b, B, c_mod, C, reuse_a, reuse_b)
  return __builtin_amdgcn_wmma_f32_16x16x32_bf16(false, a, false, b, (short)0, c,
                                                 false, false);
}

// Stage one 32x128 bf16 B-ktile (stored pre-transposed as [col][k], 8KB contiguous)
// into LDS. 128 threads, 512 x 16B chunks. Async path: 4 in-flight DMAs per wave.
__device__ __forceinline__ void stage_btile(const unsigned short* __restrict__ src,
                                            unsigned short* dst, int tid) {
#if HAVE_ASYNC_LDS
#pragma unroll
  for (int i = 0; i < 4; i++) {
    int idx = tid + i * 128;
    __builtin_amdgcn_global_load_async_to_lds_b128(
        (gv4i*)(src + idx * 8), (lv4i*)(dst + idx * 8), 0, 0);
  }
#else
  for (int i = tid; i < 512; i += 128) {
    *reinterpret_cast<us8v*>(dst + i * 8) =
        *reinterpret_cast<const us8v*>(src + i * 8);
  }
#endif
}

// load all 8 B frags from LDS then fire 8 back-to-back WMMAs
__device__ __forceinline__ void wmma_row(const unsigned short* Bsp, int nn, int half,
                                         v16bf a, v8f acc[8]) {
  v16bf bfr[8];
#pragma unroll
  for (int nt = 0; nt < 8; nt++) {
    const unsigned short* bp = &Bsp[(nt * 16 + nn) * 32 + half * 8];
    bfr[nt] = mk16(*reinterpret_cast<const us8v*>(bp),
                   *reinterpret_cast<const us8v*>(bp + 16));
  }
#pragma unroll
  for (int nt = 0; nt < 8; nt++) acc[nt] = wmma_bf16(a, bfr[nt], acc[nt]);
}

// float atomic max/min via monotone integer mapping
__device__ __forceinline__ void atomicMaxF(float* a, float v) {
  if (v >= 0.f) atomicMax((int*)a, __float_as_int(v));
  else          atomicMin((unsigned int*)a, (unsigned int)__float_as_int(v));
}
__device__ __forceinline__ void atomicMinF(float* a, float v) {
  if (v >= 0.f) atomicMin((int*)a, __float_as_int(v));
  else          atomicMax((unsigned int*)a, (unsigned int)__float_as_int(v));
}

// ---------------------------------------------------------------- utilities
__global__ void fill_f32(float* p, float v, int n) {
  int i = blockIdx.x * blockDim.x + threadIdx.x;
  if (i < n) p[i] = v;
}

// Convert a [NL][KR][128] f32 weight into bf16, pre-transposed per 32-row ktile
// as [ktile][col 0..127][k 0..31] so each ktile is a contiguous 8KB LDS image.
__global__ void cvt_wb_tr(const float* __restrict__ W, unsigned short* __restrict__ Wt,
                          int KR) {
  int i = blockIdx.x * blockDim.x + threadIdx.x;
  if (i >= NL * KR * Hh) return;
  int l = i / (KR * Hh);
  int rem = i % (KR * Hh);
  int ks = rem >> 12;       // ktile index   (4096 = 128*32)
  int t = rem & 4095;
  int c = t >> 5;           // column
  int k = t & 31;           // k within tile
  Wt[i] = f2bf(W[(size_t)l * KR * Hh + (size_t)(ks * 32 + k) * Hh + c]);
}

// ----------------------------------------------------------- atom encoder
__global__ void atom_encode(const int* __restrict__ x, const float* __restrict__ ae,
                            float* __restrict__ h, unsigned short* __restrict__ hb) {
  int i = blockIdx.x * blockDim.x + threadIdx.x;
  if (i >= NN * Hh) return;
  int node = i >> 7, c = i & 127;
  const int AOFF[9] = {0, 119, 124, 136, 148, 158, 164, 170, 172};
  float s = 0.f;
#pragma unroll
  for (int k = 0; k < 9; k++)
    s += ae[(size_t)(AOFF[k] + x[node * 9 + k]) * Hh + c];
  h[i] = s;
  hb[i] = f2bf(s);
}

// ----------------------------------------------------------------- degree
__global__ void deg_kernel(const int* __restrict__ edge_index, float* __restrict__ deg) {
  int i = blockIdx.x * blockDim.x + threadIdx.x;
  if (i >= NET) return;
  int d = (i < NE) ? edge_index[NE + i] : (i - NE);
  atomicAdd(&deg[d], 1.f);
}

__global__ void scale_kernel(float* __restrict__ deg, float* __restrict__ scale) {
  int i = blockIdx.x * blockDim.x + threadIdx.x;
  if (i >= NN) return;
  float d = fmaxf(deg[i], 1.f);
  deg[i] = d;
  scale[i] = logf(d + 1.f) * (1.f / logf(18.f));  // AVG_DEG_LOG = log 18
}

__global__ void init_reduce(float* s1, float* s2, float* mn, float* mx, int n) {
  int i = blockIdx.x * blockDim.x + threadIdx.x;
  if (i >= n) return;
  s1[i] = 0.f; s2[i] = 0.f;
  mn[i] = __builtin_bit_cast(float, 0x7F800000u);   // +inf
  mx[i] = __builtin_bit_cast(float, 0xFF800000u);   // -inf
}

__global__ void finalize_agg(float* __restrict__ s1, float* __restrict__ s2,
                             const float* __restrict__ deg, int n) {
  int i = blockIdx.x * blockDim.x + threadIdx.x;
  if (i >= n) return;
  float d = deg[i >> 7];
  float mean = s1[i] / d;
  float var = fmaxf(s2[i] / d - mean * mean, 0.f);
  s1[i] = mean;
  s2[i] = sqrtf(var + 1e-5f);
}

// ------------------------------------------ fused message GEMM + PNA reduction
// 64 edges/block, K=384: A row = [h[dst] | h[src] | e] built in registers
// (h chunks are aligned b128 loads from the L2-resident bf16 node table,
//  e = bond-embedding sum computed inline). B ktiles double-buffered in LDS
//  via async DMA. Epilogue reduces m into sum/sumsq/min/max with atomics.
__global__ __launch_bounds__(128) void gemm1_msg(
    const unsigned short* __restrict__ hb, const int* __restrict__ edge_index,
    const int* __restrict__ edge_attr, const float* __restrict__ bond_emb,
    const unsigned short* __restrict__ Wt, const float* __restrict__ bias,
    float* __restrict__ s1, float* __restrict__ s2,
    float* __restrict__ mn, float* __restrict__ mx) {
  __shared__ unsigned short Bs[2][32 * 128];
  __shared__ int srcS[64], dstS[64];
  const int tid = threadIdx.x;
  const int eb = blockIdx.x * 64;
  const int lane = tid & 31, wave = tid >> 5;
  const int half = lane >> 4, nn = lane & 15;
  const int arow = wave * 16 + nn;
  __builtin_prefetch(&edge_index[eb], 0, 3);

  for (int r = tid; r < 64; r += 128) {
    int e = eb + r, s, d;
    if (e < NE) { s = edge_index[e]; d = edge_index[NE + e]; }
    else        { s = d = e - NE; }
    srcS[r] = s; dstS[r] = d;
  }

  // per-lane bond-embedding rows for this lane's A row (uniform e<NE per block)
  const int eMine = eb + arow;
  int j0 = 0, j1 = 5, j2 = 11;
  if (eMine < NE) {
    j0 = edge_attr[eMine * 3 + 0];
    j1 = 5 + edge_attr[eMine * 3 + 1];
    j2 = 11 + edge_attr[eMine * 3 + 2];
  }
  const float* bz0 = bond_emb + (size_t)j0 * Hh;
  const float* bz1 = bond_emb + (size_t)j1 * Hh;
  const float* bz2 = bond_emb + (size_t)j2 * Hh;

  __syncthreads();  // srcS/dstS visible
  const unsigned short* hd = hb + (size_t)dstS[arow] * Hh;
  const unsigned short* hs = hb + (size_t)srcS[arow] * Hh;

  v8f acc[8] = {};
  stage_btile(Wt, Bs[0], tid);
  int p = 0;
  for (int ks = 0; ks < 12; ks++) {
    if (ks + 1 < 12) {
      stage_btile(Wt + (ks + 1) * 4096, Bs[p ^ 1], tid);
      WAIT_ASYNC(4);          // retire tile ks, keep tile ks+1 in flight
    } else {
      WAIT_ASYNC(0);
    }
    __syncthreads();

    v16bf a;
    if (ks < 8) {
      const unsigned short* src = (ks < 4) ? hd + ks * 32 : hs + (ks - 4) * 32;
      a = mk16(*reinterpret_cast<const us8v*>(src + half * 8),
               *reinterpret_cast<const us8v*>(src + 16 + half * 8));
    } else {
      int cb = (ks - 8) * 32;
      us8v lo, hi;
#pragma unroll
      for (int j = 0; j < 8; j++) {
        int c0 = cb + half * 8 + j;
        int c1 = cb + 16 + half * 8 + j;
        lo[j] = f2bf(bz0[c0] + bz1[c0] + bz2[c0]);
        hi[j] = f2bf(bz0[c1] + bz1[c1] + bz2[c1]);
      }
      a = mk16(lo, hi);
    }
    wmma_row(Bs[p], nn, half, a, acc);
    __syncthreads();          // reads of Bs[p] done before it is re-staged
    p ^= 1;
  }

  // epilogue: m = acc + bias; reduce into per-dst sum/sumsq/min/max
#pragma unroll
  for (int nt = 0; nt < 8; nt++) {
    int col = nt * 16 + nn;
    float bv = bias[col];
#pragma unroll
    for (int r = 0; r < 8; r++) {
      int row = wave * 16 + r + half * 8;
      float mval = acc[nt][r] + bv;
      size_t o = (size_t)dstS[row] * Hh + col;
      atomicAdd(&s1[o], mval);
      atomicAdd(&s2[o], mval * mval);
      atomicMaxF(&mx[o], mval);
      atomicMinF(&mn[o], mval);
    }
  }
}

// -------------------- post GEMM: [64 nodes, 13H] @ Wpost, scalers fused in regs
__global__ __launch_bounds__(128) void gemm2_post(
    const unsigned short* __restrict__ hb, const float* __restrict__ mean_,
    const float* __restrict__ std_, const float* __restrict__ mn,
    const float* __restrict__ mx, const float* __restrict__ scale,
    const unsigned short* __restrict__ Wt, const float* __restrict__ bias,
    float* __restrict__ out) {
  __shared__ unsigned short Bs[2][32 * 128];
  const int tid = threadIdx.x;
  const int nb = blockIdx.x * 64;
  const int lane = tid & 31, wave = tid >> 5;
  const int half = lane >> 4, nn = lane & 15;
  const int arow = wave * 16 + nn;
  const int row = nb + arow;
  const float scl = scale[row];
  const float inv = 1.f / scl;

  v8f acc[8] = {};
  stage_btile(Wt, Bs[0], tid);
  int p = 0;
  for (int ks = 0; ks < 52; ks++) {
    if (ks + 1 < 52) {
      stage_btile(Wt + (size_t)(ks + 1) * 4096, Bs[p ^ 1], tid);
      WAIT_ASYNC(4);
    } else {
      WAIT_ASYNC(0);
    }
    __syncthreads();

    int kb = ks >> 2, cb = (ks & 3) * 32;
    v16bf a;
    if (kb == 0) {
      const unsigned short* src = hb + (size_t)row * Hh + cb;
      a = mk16(*reinterpret_cast<const us8v*>(src + half * 8),
               *reinterpret_cast<const us8v*>(src + 16 + half * 8));
    } else {
      int q = (kb - 1) & 3, sc = (kb - 1) >> 2;
      const float* src = (q == 0) ? mean_ : (q == 1) ? mn : (q == 2) ? mx : std_;
      src += (size_t)row * Hh + cb;
      float fac = (sc == 0) ? 1.f : (sc == 1) ? scl : inv;
      us8v lo, hi;
#pragma unroll
      for (int j = 0; j < 8; j++) {
        lo[j] = f2bf(src[half * 8 + j] * fac);
        hi[j] = f2bf(src[16 + half * 8 + j] * fac);
      }
      a = mk16(lo, hi);
    }
    wmma_row(Bs[p], nn, half, a, acc);
    __syncthreads();
    p ^= 1;
  }
#pragma unroll
  for (int nt = 0; nt < 8; nt++) {
    int col = nt * 16 + nn;
    float bv = bias[col];
#pragma unroll
    for (int r = 0; r < 8; r++) {
      int orow = nb + wave * 16 + r + half * 8;
      out[(size_t)orow * Hh + col] = acc[nt][r] + bv;
    }
  }
}

// --------------------------------------------------- lin GEMM: [64,128]@[128,128]
__global__ __launch_bounds__(128) void gemm3_lin(
    const float* __restrict__ X, const unsigned short* __restrict__ Wt,
    const float* __restrict__ bias, float* __restrict__ out) {
  __shared__ unsigned short Bs[2][32 * 128];
  const int tid = threadIdx.x;
  const int nb = blockIdx.x * 64;
  const int lane = tid & 31, wave = tid >> 5;
  const int half = lane >> 4, nn = lane & 15;
  const int arow = wave * 16 + nn;
  const float* xr = X + (size_t)(nb + arow) * Hh;

  v8f acc[8] = {};
  stage_btile(Wt, Bs[0], tid);
  int p = 0;
  for (int ks = 0; ks < 4; ks++) {
    if (ks + 1 < 4) {
      stage_btile(Wt + (ks + 1) * 4096, Bs[p ^ 1], tid);
      WAIT_ASYNC(4);
    } else {
      WAIT_ASYNC(0);
    }
    __syncthreads();
    us8v lo, hi;
#pragma unroll
    for (int j = 0; j < 8; j++) {
      lo[j] = f2bf(xr[ks * 32 + half * 8 + j]);
      hi[j] = f2bf(xr[ks * 32 + 16 + half * 8 + j]);
    }
    v16bf a = mk16(lo, hi);
    wmma_row(Bs[p], nn, half, a, acc);
    __syncthreads();
    p ^= 1;
  }
#pragma unroll
  for (int nt = 0; nt < 8; nt++) {
    int col = nt * 16 + nn;
    float bv = bias[col];
#pragma unroll
    for (int r = 0; r < 8; r++) {
      int orow = nb + wave * 16 + r + half * 8;
      out[(size_t)orow * Hh + col] = acc[nt][r] + bv;
    }
  }
}

// -------------------------------------------------------- batchnorm / relu / pool
__global__ void colstats(const float* __restrict__ X, float* __restrict__ mu,
                         float* __restrict__ ss, int rows, int cols) {
  int i = blockIdx.x * blockDim.x + threadIdx.x;
  if (i >= rows * cols) return;
  float v = X[i];
  int c = i % cols;
  atomicAdd(&mu[c], v);
  atomicAdd(&ss[c], v * v);
}

__global__ void bn_relu(const float* __restrict__ X, const float* __restrict__ mu,
                        const float* __restrict__ ss, const float* __restrict__ gm,
                        const float* __restrict__ bt, float* __restrict__ Y,
                        unsigned short* __restrict__ Yb, int rows, int cols) {
  int i = blockIdx.x * blockDim.x + threadIdx.x;
  if (i >= rows * cols) return;
  int c = i % cols;
  float inv = 1.f / (float)rows;
  float mean = mu[c] * inv;
  float var = ss[c] * inv - mean * mean;
  float z = (X[i] - mean) * rsqrtf(var + 1e-5f) * gm[c] + bt[c];
  z = fmaxf(z, 0.f);
  Y[i] = z;
  if (Yb) Yb[i] = f2bf(z);
}

__global__ void pool_kernel(const float* __restrict__ h, const int* __restrict__ batch,
                            float* __restrict__ g) {
  int i = blockIdx.x * blockDim.x + threadIdx.x;
  if (i >= NN * Hh) return;
  int node = i >> 7, c = i & 127;
  atomicAdd(&g[(size_t)batch[node] * Hh + c], h[i]);
}

__global__ void dense(const float* __restrict__ X, const float* __restrict__ W,
                      const float* __restrict__ b, float* __restrict__ Y,
                      int rows, int K, int cols) {
  int i = blockIdx.x * blockDim.x + threadIdx.x;
  if (i >= rows * cols) return;
  int r = i / cols, c = i % cols;
  float s = b[c];
  for (int k = 0; k < K; k++) s += X[(size_t)r * K + k] * W[(size_t)k * cols + c];
  Y[i] = s;
}

// ---------------------------------------------------------------------- launch
extern "C" void kernel_launch(void* const* d_in, const int* in_sizes, int n_in,
                              void* d_out, int out_size, void* d_ws, size_t ws_size,
                              hipStream_t stream) {
  (void)in_sizes; (void)n_in; (void)out_size; (void)ws_size;
  const int*   x          = (const int*)d_in[0];
  const int*   edge_index = (const int*)d_in[1];
  const int*   edge_attr  = (const int*)d_in[2];
  const int*   batch      = (const int*)d_in[3];
  const float* atom_emb   = (const float*)d_in[4];
  const float* bond_emb   = (const float*)d_in[5];
  const float* Wpre  = (const float*)d_in[6];
  const float* bpre  = (const float*)d_in[7];
  const float* Wpost = (const float*)d_in[8];
  const float* bpost = (const float*)d_in[9];
  const float* Wlin  = (const float*)d_in[10];
  const float* blin  = (const float*)d_in[11];
  const float* bng   = (const float*)d_in[12];
  const float* bnb   = (const float*)d_in[13];
  const float* W1 = (const float*)d_in[14]; const float* b1  = (const float*)d_in[15];
  const float* g1 = (const float*)d_in[16]; const float* be1 = (const float*)d_in[17];
  const float* W2 = (const float*)d_in[18]; const float* b2  = (const float*)d_in[19];
  const float* g2 = (const float*)d_in[20]; const float* be2 = (const float*)d_in[21];
  const float* W3 = (const float*)d_in[22]; const float* b3  = (const float*)d_in[23];
  float* out = (float*)d_out;

  char* base = (char*)d_ws;
  size_t off = 0;
  auto wsalloc = [&](size_t bytes) -> void* {
    void* p = base + off;
    off += (bytes + 255) & ~(size_t)255;
    return p;
  };
  float*          h      = (float*)wsalloc((size_t)NN * Hh * 4);
  unsigned short* hb     = (unsigned short*)wsalloc((size_t)NN * Hh * 2);
  float*          s1     = (float*)wsalloc((size_t)NN * Hh * 4);
  float*          s2     = (float*)wsalloc((size_t)NN * Hh * 4);
  float*          mn     = (float*)wsalloc((size_t)NN * Hh * 4);
  float*          mx     = (float*)wsalloc((size_t)NN * Hh * 4);
  float*          degb   = (float*)wsalloc((size_t)NN * 4);
  float*          scaleb = (float*)wsalloc((size_t)NN * 4);
  float*          out1   = (float*)wsalloc((size_t)NN * Hh * 4);
  float*          out2   = (float*)wsalloc((size_t)NN * Hh * 4);
  unsigned short* WpreT  = (unsigned short*)wsalloc((size_t)NL * 384 * 128 * 2);
  unsigned short* WpostT = (unsigned short*)wsalloc((size_t)NL * 1664 * 128 * 2);
  unsigned short* WlinT  = (unsigned short*)wsalloc((size_t)NL * 128 * 128 * 2);
  float*          gpool  = (float*)wsalloc((size_t)NG * Hh * 4);
  float*          z1     = (float*)wsalloc((size_t)NG * 64 * 4);
  float*          z1r    = (float*)wsalloc((size_t)NG * 64 * 4);
  float*          z2     = (float*)wsalloc((size_t)NG * 32 * 4);
  float*          z2r    = (float*)wsalloc((size_t)NG * 32 * 4);
  float*          stats  = (float*)wsalloc(2 * 128 * 4);
  float* mu = stats;
  float* ssb = stats + 128;

  const int TB = 256;
  auto nblk = [](long n, int tb) { return (unsigned)((n + tb - 1) / tb); };

  cvt_wb_tr<<<nblk((long)NL * 384 * 128, TB), TB, 0, stream>>>(Wpre, WpreT, 384);
  cvt_wb_tr<<<nblk((long)NL * 1664 * 128, TB), TB, 0, stream>>>(Wpost, WpostT, 1664);
  cvt_wb_tr<<<nblk((long)NL * 128 * 128, TB), TB, 0, stream>>>(Wlin, WlinT, 128);

  atom_encode<<<nblk((long)NN * Hh, TB), TB, 0, stream>>>(x, atom_emb, h, hb);

  fill_f32<<<nblk(NN, TB), TB, 0, stream>>>(degb, 0.f, NN);
  deg_kernel<<<nblk(NET, TB), TB, 0, stream>>>(edge_index, degb);
  scale_kernel<<<nblk(NN, TB), TB, 0, stream>>>(degb, scaleb);

  for (int l = 0; l < NL; l++) {
    init_reduce<<<nblk((long)NN * Hh, TB), TB, 0, stream>>>(s1, s2, mn, mx, NN * Hh);
    gemm1_msg<<<NET / 64, 128, 0, stream>>>(hb, edge_index, edge_attr, bond_emb,
                                            WpreT + (size_t)l * 384 * 128,
                                            bpre + l * Hh, s1, s2, mn, mx);
    finalize_agg<<<nblk((long)NN * Hh, TB), TB, 0, stream>>>(s1, s2, degb, NN * Hh);
    gemm2_post<<<NN / 64, 128, 0, stream>>>(hb, s1, s2, mn, mx, scaleb,
                                            WpostT + (size_t)l * 1664 * 128,
                                            bpost + l * Hh, out1);
    gemm3_lin<<<NN / 64, 128, 0, stream>>>(out1, WlinT + (size_t)l * 128 * 128,
                                           blin + l * Hh, out2);
    fill_f32<<<1, 256, 0, stream>>>(stats, 0.f, 256);
    colstats<<<nblk((long)NN * Hh, TB), TB, 0, stream>>>(out2, mu, ssb, NN, Hh);
    bn_relu<<<nblk((long)NN * Hh, TB), TB, 0, stream>>>(out2, mu, ssb, bng + l * Hh,
                                                        bnb + l * Hh, h, hb, NN, Hh);
  }

  fill_f32<<<nblk((long)NG * Hh, TB), TB, 0, stream>>>(gpool, 0.f, NG * Hh);
  pool_kernel<<<nblk((long)NN * Hh, TB), TB, 0, stream>>>(h, batch, gpool);

  dense<<<nblk(NG * 64, TB), TB, 0, stream>>>(gpool, W1, b1, z1, NG, 128, 64);
  fill_f32<<<1, 256, 0, stream>>>(stats, 0.f, 256);
  colstats<<<nblk(NG * 64, TB), TB, 0, stream>>>(z1, mu, ssb, NG, 64);
  bn_relu<<<nblk(NG * 64, TB), TB, 0, stream>>>(z1, mu, ssb, g1, be1, z1r,
                                                (unsigned short*)nullptr, NG, 64);

  dense<<<nblk(NG * 32, TB), TB, 0, stream>>>(z1r, W2, b2, z2, NG, 64, 32);
  fill_f32<<<1, 256, 0, stream>>>(stats, 0.f, 256);
  colstats<<<nblk(NG * 32, TB), TB, 0, stream>>>(z2, mu, ssb, NG, 32);
  bn_relu<<<nblk(NG * 32, TB), TB, 0, stream>>>(z2, mu, ssb, g2, be2, z2r,
                                                (unsigned short*)nullptr, NG, 32);

  dense<<<nblk(NG * NP, TB), TB, 0, stream>>>(z2r, W3, b3, out, NG, 32, NP);
}
